// Dehazeblock_49873160241354
// MI455X (gfx1250) — matrix operand order repaired and checked by reading, compiled, not run on hardware
//
#include <hip/hip_runtime.h>
#include <cstdint>
#include <cstddef>

// ---------------------------------------------------------------------------
// Problem constants
// ---------------------------------------------------------------------------
#define BATCH   4
#define DIM     256
#define HH      64
#define WW      64
#define HP      66          // padded
#define WP      66
#define NTAP    9
#define KTOT    (NTAP * DIM)          // 2304
#define NPIX    (BATCH * HH * WW)     // 16384
#define NCHUNK  (KTOT / 32)           // 72 K-chunks of 32
#define BSTRIDE 40                    // padded LDS row stride (bf16 elems, 80 B)

typedef __bf16 bf16_t;
typedef __attribute__((ext_vector_type(16))) __bf16 v16bf;
typedef __attribute__((ext_vector_type(8)))  float  v8f;

// ---------------------------------------------------------------------------
// Fragment helpers (CDNA5 wave32 WMMA layouts, CDNA5 ISA)
//  A 16x32 bf16 : lane L -> M = L%16 ; elems e     = K (L/16)*16 + e
//  B 32x16 bf16 : lane L -> N = L%16 ; elems e = K (L/16)*16 + e (this code's contiguous order)
// ---------------------------------------------------------------------------
__device__ __forceinline__ v16bf load_frag2(const bf16_t* p0, const bf16_t* p1) {
    union { v16bf v; float4 f[2]; } u;
    u.f[0] = *reinterpret_cast<const float4*>(p0);
    u.f[1] = *reinterpret_cast<const float4*>(p1);
    return u.v;
}

__device__ __forceinline__ v8f wmma_bf16(v16bf a, v16bf b, v8f c) {
    return __builtin_amdgcn_wmma_f32_16x16x32_bf16(
        /*neg_a*/false, a, /*neg_b*/false, b,
        /*c_mod*/(short)0, c, /*reuse_a*/false, /*reuse_b*/false);
}

__device__ __forceinline__ v8f zero_v8f() {
    v8f z = {0.f, 0.f, 0.f, 0.f, 0.f, 0.f, 0.f, 0.f};
    return z;
}

// Low 32 bits of a generic pointer into __shared__ == LDS byte offset
// (generic LDS addresses carry the AS(3) offset in addr[31:0]).
__device__ __forceinline__ uint32_t lds_off32(const void* p) {
    return (uint32_t)(uintptr_t)p;
}

// Async copy of 16 bytes global -> LDS (GLOBAL_LOAD_ASYNC_TO_LDS_B128,
// GV mode, tracked by ASYNCcnt).  CDNA5 ISA.
__device__ __forceinline__ void async_g2l_b128(uint32_t lds_byte_off,
                                               const void* gaddr) {
    asm volatile("global_load_async_to_lds_b128 %0, %1, off"
                 :
                 : "v"(lds_byte_off), "v"((uint64_t)(uintptr_t)gaddr)
                 : "memory");
}

__device__ __forceinline__ void wait_async_le8() {
    asm volatile("s_wait_asynccnt 0x8" ::: "memory");
}
__device__ __forceinline__ void wait_async_0() {
    asm volatile("s_wait_asynccnt 0x0" ::: "memory");
}

// ---------------------------------------------------------------------------
// K0a: zero a dword range (padded NHWC intermediates' halos)
// ---------------------------------------------------------------------------
__global__ void zero_dwords(uint32_t* __restrict__ p, int n) {
    int i = blockIdx.x * blockDim.x + threadIdx.x;
    if (i < n) p[i] = 0u;
}

// ---------------------------------------------------------------------------
// K0b: fp32 NCHW -> bf16 padded NHWC  [B][66][66][256]
// ---------------------------------------------------------------------------
__global__ void pack_x(const float* __restrict__ x, bf16_t* __restrict__ xh) {
    int idx = blockIdx.x * blockDim.x + threadIdx.x;   // over B*66*66*256
    int ci   = idx & 255;
    int rest = idx >> 8;
    int col  = rest % WP;  rest /= WP;
    int row  = rest % HP;
    int b    = rest / HP;
    float v = 0.f;
    if (row >= 1 && row <= HH && col >= 1 && col <= WW)
        v = x[(((size_t)b * DIM + ci) * HH + (row - 1)) * WW + (col - 1)];
    xh[idx] = (bf16_t)v;
}

// ---------------------------------------------------------------------------
// K0c: OIHW fp32 -> bf16 [co_pad][tap*256+ci]
// ---------------------------------------------------------------------------
__global__ void pack_w(const float* __restrict__ w, bf16_t* __restrict__ wt,
                       int co_real, int co_pad) {
    int idx = blockIdx.x * blockDim.x + threadIdx.x;   // over co_pad*KTOT
    int k   = idx % KTOT;
    int row = idx / KTOT;
    int t   = k / DIM;
    int ci  = k % DIM;
    float v = (row < co_real) ? w[((size_t)row * DIM + ci) * NTAP + t] : 0.f;
    wt[idx] = (bf16_t)v;
}

// ---------------------------------------------------------------------------
// Implicit-GEMM 3x3 conv (pad=1), M=16384 pixels, N=256, K=2304.
// Block = 128 thr = 4 waves; wave (w>>1) owns a 16-pixel M tile, wave (w&1)
// selects a 128-wide N half (8 accumulator tiles).
// The 256x32 B chunk-tile is double-buffered in LDS and staged with
// GLOBAL_LOAD_ASYNC_TO_LDS_B128 (ASYNCcnt) while WMMAs consume the previous
// chunk.  A fragments stream straight from global (padded NHWC, L2-resident).
// MODE 1: out = relu(acc + bias) + residual(NCHW fp32)  -> padded NHWC bf16
// MODE 2: out = acc + bias                              -> padded NHWC bf16
// ---------------------------------------------------------------------------
template <int MODE>
__global__ __launch_bounds__(128, 1)
void conv_gemm(const bf16_t* __restrict__ inp,     // [B][66][66][256] bf16
               const bf16_t* __restrict__ wt,      // [256][2304] bf16
               const float*  __restrict__ bias,    // [256]
               const float*  __restrict__ xres,    // NCHW fp32 (MODE 1)
               bf16_t*       __restrict__ outp)    // [B][66][66][256] bf16
{
    __shared__ __align__(16) bf16_t Bs[2][DIM * BSTRIDE];   // 2 x 20 KB

    const int tid   = threadIdx.x;
    const int lane  = tid & 31;
    const int wave  = tid >> 5;                    // 0..3
    const int lanem = lane & 15;
    const int half  = lane >> 4;

    const int mtile = blockIdx.x * 32 + (wave >> 1) * 16;
    const int n0    = (wave & 1) * 128;

    const int bidx  = mtile >> 12;                 // /4096
    const int hrow  = (mtile >> 6) & 63;
    const int wbase = mtile & 63;                  // multiple of 16

    // ---- async stage of B chunk-tile c into LDS buffer `buf` -------------
    // 256 rows x 64 B; 1024 b128 transfers; 8 per thread.
    auto stage_b = [&](int buf, int c) {
        const int kb = c * 32;                     // K base (elements)
#pragma unroll
        for (int q = 0; q < 8; ++q) {
            const int idx = q * 128 + tid;         // 0..1023
            const int row = idx >> 2;
            const int seg = idx & 3;               // 4 x 16B per row
            const bf16_t* g = wt + (size_t)row * KTOT + kb + seg * 8;
            const uint32_t l = lds_off32(&Bs[buf][row * BSTRIDE + seg * 8]);
            async_g2l_b128(l, g);
        }
    };

    v8f acc[8];
#pragma unroll
    for (int i = 0; i < 8; ++i) acc[i] = zero_v8f();

    stage_b(0, 0);

    for (int c = 0; c < NCHUNK; ++c) {
        const int buf = c & 1;
        if (c + 1 < NCHUNK) {
            stage_b(buf ^ 1, c + 1);
            wait_async_le8();                      // chunk c's 8 loads landed
        } else {
            wait_async_0();
        }
        __syncthreads();                           // all waves' stages landed

        // A fragment: tap/channel decode of this K chunk
        const int t  = c >> 3;                     // tap 0..8
        const int k0 = (c & 7) * 32;               // channel base within tap
        const int ty = t / 3, tx = t % 3;
        const bf16_t* arow = inp +
            ((size_t)((bidx * HP + hrow + ty) * WP + (wbase + lanem + tx))) * DIM;
        v16bf afrag = load_frag2(arow + k0 + half * 16,
                                 arow + k0 + half * 16 + 8);

#pragma unroll
        for (int nt = 0; nt < 8; ++nt) {
            const bf16_t* brow =
                &Bs[buf][(n0 + nt * 16 + lanem) * BSTRIDE + half * 16];
            v16bf bfrag = load_frag2(brow, brow + 8);
            acc[nt] = wmma_bf16(afrag, bfrag, acc[nt]);
        }
        __syncthreads();                           // done reading buf
    }

#pragma unroll
    for (int nt = 0; nt < 8; ++nt) {
        const int n  = n0 + nt * 16 + lanem;
        const float bn = bias[n];
#pragma unroll
        for (int v = 0; v < 8; ++v) {
            const int wx = wbase + v + half * 8;   // pixel column (M = v + half*8)
            float val = acc[nt][v] + bn;
            if (MODE == 1) {
                val = fmaxf(val, 0.f) +
                      xres[(((size_t)bidx * DIM + n) * HH + hrow) * WW + wx];
            }
            outp[((size_t)(bidx * HP + hrow + 1) * WP + (wx + 1)) * DIM + n] =
                (bf16_t)val;
        }
    }
}

// ---------------------------------------------------------------------------
// Offset conv: N=32 (18 real rows of p_w, zero-padded).  Writes fp32
// offsets as [pixel][18].  Small (N=32) -> direct-global B fragments.
// ---------------------------------------------------------------------------
__global__ __launch_bounds__(128, 1)
void offset_conv(const bf16_t* __restrict__ y2p,   // [B][66][66][256]
                 const bf16_t* __restrict__ pwt,   // [32][2304]
                 const float*  __restrict__ pb,    // [18]
                 float*        __restrict__ off)   // [16384][18]
{
    const int lane  = threadIdx.x & 31;
    const int wave  = threadIdx.x >> 5;
    const int lanem = lane & 15;
    const int half  = lane >> 4;

    const int mtile = blockIdx.x * 64 + wave * 16;
    const int bidx  = mtile >> 12;
    const int hrow  = (mtile >> 6) & 63;
    const int wbase = mtile & 63;

    v8f acc[2] = {zero_v8f(), zero_v8f()};

    for (int t = 0; t < NTAP; ++t) {
        const int ty = t / 3, tx = t % 3;
        const bf16_t* arow = y2p +
            ((size_t)((bidx * HP + hrow + ty) * WP + (wbase + lanem + tx))) * DIM;
        for (int cc = 0; cc < 8; ++cc) {
            const int k0 = cc * 32;
            v16bf afrag = load_frag2(arow + k0 + half * 16,
                                     arow + k0 + half * 16 + 8);
            const int kb = t * DIM + k0 + half * 16;
#pragma unroll
            for (int nt = 0; nt < 2; ++nt) {
                const bf16_t* brow = pwt + (size_t)(nt * 16 + lanem) * KTOT + kb;
                v16bf bfrag = load_frag2(brow, brow + 8);
                acc[nt] = wmma_bf16(afrag, bfrag, acc[nt]);
            }
        }
    }

#pragma unroll
    for (int nt = 0; nt < 2; ++nt) {
        const int n = nt * 16 + lanem;
        if (n < 18) {
            const float bn = pb[n];
#pragma unroll
            for (int v = 0; v < 8; ++v) {
                const int m = mtile + v + half * 8;
                off[(size_t)m * 18 + n] = acc[nt][v] + bn;
            }
        }
    }
}

// ---------------------------------------------------------------------------
// Deformable sampling + final GEMM.
// Workgroup = 256 thr (8 waves) owns one 16-pixel M tile.
// For each of 9 taps: build the bilinear-gathered A tile (16x256 bf16) in
// LDS, then every wave accumulates its 32 output channels via WMMA.
// Epilogue: += residual x (fp32 NCHW), store fp32 NCHW.
// ---------------------------------------------------------------------------
__global__ __launch_bounds__(256, 1)
void deform_gemm(const bf16_t* __restrict__ y2p,   // [B][66][66][256]
                 const bf16_t* __restrict__ dwt,   // [256][2304]
                 const float*  __restrict__ off,   // [16384][18]
                 const float*  __restrict__ xres,  // NCHW fp32
                 float*        __restrict__ out)   // NCHW fp32
{
    __shared__ __align__(16) bf16_t As[16 * DIM];  // 8 KB

    const int lane  = threadIdx.x & 31;
    const int wave  = threadIdx.x >> 5;            // 0..7 -> N range wave*32
    const int lanem = lane & 15;
    const int half  = lane >> 4;

    const int mtile = blockIdx.x * 16;
    const int bidx  = mtile >> 12;
    const int hrow  = (mtile >> 6) & 63;
    const int wbase = mtile & 63;

    v8f acc[2] = {zero_v8f(), zero_v8f()};

    // phase-1 work assignment: thread -> (pixel, 16-channel segment)
    const int p1_pix  = threadIdx.x >> 4;          // 0..15
    const int p1_cseg = (threadIdx.x & 15) * 16;   // 0,16,..,240

    for (int t = 0; t < NTAP; ++t) {
        // ---- phase 1: bilinear gather into LDS A tile --------------------
        {
            const int m  = mtile + p1_pix;
            const int wx = wbase + p1_pix;
            const float ox = off[(size_t)m * 18 + t];
            const float oy = off[(size_t)m * 18 + 9 + t];
            const float px = ox + (float)(t / 3 - 1) + (float)(hrow + 1);
            const float py = oy + (float)(t % 3 - 1) + (float)(wx + 1);
            const float fx = floorf(px), fy = floorf(py);
            const float x0 = fminf(fmaxf(fx,       0.f), 65.f);
            const float x1 = fminf(fmaxf(fx + 1.f, 0.f), 65.f);
            const float y0 = fminf(fmaxf(fy,       0.f), 65.f);
            const float y1 = fminf(fmaxf(fy + 1.f, 0.f), 65.f);
            const float pxc = fminf(fmaxf(px, 0.f), 65.f);
            const float pyc = fminf(fmaxf(py, 0.f), 65.f);
            const float glt = (1.f + x0 - pxc) * (1.f + y0 - pyc);
            const float grb = (1.f - x1 + pxc) * (1.f - y1 + pyc);
            const float glb = (1.f + x0 - pxc) * (1.f - y1 + pyc);
            const float grt = (1.f - x1 + pxc) * (1.f + y0 - pyc);
            const int ix0 = (int)x0, ix1 = (int)x1, iy0 = (int)y0, iy1 = (int)y1;

            const bf16_t* base = y2p + (size_t)bidx * HP * WP * DIM + p1_cseg;
            const bf16_t* plt  = base + (size_t)(ix0 * WP + iy0) * DIM;
            const bf16_t* prb  = base + (size_t)(ix1 * WP + iy1) * DIM;
            const bf16_t* plb  = base + (size_t)(ix0 * WP + iy1) * DIM;
            const bf16_t* prt  = base + (size_t)(ix1 * WP + iy0) * DIM;

            union { float4 f[2]; bf16_t h[16]; } clt, crb, clb, crt, res;
            clt.f[0] = ((const float4*)plt)[0]; clt.f[1] = ((const float4*)plt)[1];
            crb.f[0] = ((const float4*)prb)[0]; crb.f[1] = ((const float4*)prb)[1];
            clb.f[0] = ((const float4*)plb)[0]; clb.f[1] = ((const float4*)plb)[1];
            crt.f[0] = ((const float4*)prt)[0]; crt.f[1] = ((const float4*)prt)[1];
#pragma unroll
            for (int c = 0; c < 16; ++c) {
                float v = glt * (float)clt.h[c] + grb * (float)crb.h[c] +
                          glb * (float)clb.h[c] + grt * (float)crt.h[c];
                res.h[c] = (bf16_t)v;
            }
            float4* dst = (float4*)&As[p1_pix * DIM + p1_cseg];
            dst[0] = res.f[0];
            dst[1] = res.f[1];
        }
        __syncthreads();

        // ---- phase 2: WMMA over this tap's 256-wide K slab ---------------
        for (int cc = 0; cc < 8; ++cc) {
            const bf16_t* ap = &As[lanem * DIM + cc * 32 + half * 16];
            v16bf afrag = load_frag2(ap, ap + 8);
            const int kb = t * DIM + cc * 32 + half * 16;
#pragma unroll
            for (int nt = 0; nt < 2; ++nt) {
                const bf16_t* brow =
                    dwt + (size_t)(wave * 32 + nt * 16 + lanem) * KTOT + kb;
                v16bf bfrag = load_frag2(brow, brow + 8);
                acc[nt] = wmma_bf16(afrag, bfrag, acc[nt]);
            }
        }
        __syncthreads();
    }

    // ---- epilogue: residual add, NCHW fp32 store -------------------------
#pragma unroll
    for (int nt = 0; nt < 2; ++nt) {
        const int n = wave * 32 + nt * 16 + lanem;
#pragma unroll
        for (int v = 0; v < 8; ++v) {
            const int wx = wbase + v + half * 8;
            const size_t idx = (((size_t)bidx * DIM + n) * HH + hrow) * WW + wx;
            out[idx] = acc[nt][v] + xres[idx];
        }
    }
}

// ---------------------------------------------------------------------------
// Host-side orchestration
// ---------------------------------------------------------------------------
extern "C" void kernel_launch(void* const* d_in, const int* in_sizes, int n_in,
                              void* d_out, int out_size, void* d_ws, size_t ws_size,
                              hipStream_t stream) {
    const float* x   = (const float*)d_in[0];
    const float* w1  = (const float*)d_in[1];
    const float* b1  = (const float*)d_in[2];
    const float* w2  = (const float*)d_in[3];
    const float* b2  = (const float*)d_in[4];
    const float* p_w = (const float*)d_in[5];
    const float* p_b = (const float*)d_in[6];
    const float* dw  = (const float*)d_in[7];

    const size_t SZP  = (size_t)BATCH * HP * WP * DIM * sizeof(bf16_t); // 8,921,088
    const size_t SZW  = (size_t)DIM * KTOT * sizeof(bf16_t);            // 1,179,648
    const size_t SZPW = (size_t)32 * KTOT * sizeof(bf16_t);             //   147,456
    const size_t SZOFF = (size_t)NPIX * 18 * sizeof(float);             // 1,179,648

    const size_t OFF_XH  = 0;
    const size_t OFF_Y1  = OFF_XH + SZP;
    const size_t OFF_Y2  = OFF_Y1 + SZP;
    const size_t OFF_W1T = OFF_Y2 + SZP;
    const size_t OFF_W2T = OFF_W1T + SZW;
    const size_t OFF_DWT = OFF_W2T + SZW;
    const size_t OFF_PWT = OFF_DWT + SZW;
    const size_t OFF_OFS = OFF_PWT + SZPW;
    const size_t NEEDED  = OFF_OFS + SZOFF;            // ~30.2 MB
    if (ws_size < NEEDED) return;

    char* ws = (char*)d_ws;
    bf16_t* xh  = (bf16_t*)(ws + OFF_XH);
    bf16_t* y1p = (bf16_t*)(ws + OFF_Y1);
    bf16_t* y2p = (bf16_t*)(ws + OFF_Y2);
    bf16_t* w1t = (bf16_t*)(ws + OFF_W1T);
    bf16_t* w2t = (bf16_t*)(ws + OFF_W2T);
    bf16_t* dwt = (bf16_t*)(ws + OFF_DWT);
    bf16_t* pwt = (bf16_t*)(ws + OFF_PWT);
    float*  ofs = (float*)(ws + OFF_OFS);

    const int NPAD_ELEMS = BATCH * HP * WP * DIM;      // 4,460,544 bf16 per buffer

    // zero y1p + y2p (contiguous): 2 bf16 buffers == NPAD_ELEMS dwords total
    zero_dwords<<<NPAD_ELEMS / 256, 256, 0, stream>>>((uint32_t*)y1p, NPAD_ELEMS);

    pack_x<<<NPAD_ELEMS / 256, 256, 0, stream>>>(x, xh);
    pack_w<<<(DIM * KTOT) / 256, 256, 0, stream>>>(w1, w1t, DIM, DIM);
    pack_w<<<(DIM * KTOT) / 256, 256, 0, stream>>>(w2, w2t, DIM, DIM);
    pack_w<<<(DIM * KTOT) / 256, 256, 0, stream>>>(dw, dwt, DIM, DIM);
    pack_w<<<(32 * KTOT) / 256, 256, 0, stream>>>(p_w, pwt, 18, 32);

    // conv1: relu(conv(x,w1)+b1) + x  -> y1p (padded NHWC bf16)
    conv_gemm<1><<<NPIX / 32, 128, 0, stream>>>(xh, w1t, b1, x, y1p);
    // conv2: conv(y1,w2)+b2 -> y2p
    conv_gemm<2><<<NPIX / 32, 128, 0, stream>>>(y1p, w2t, b2, nullptr, y2p);
    // offsets: conv(y2, p_w)+p_b -> ofs[pixel][18]
    offset_conv<<<NPIX / 64, 128, 0, stream>>>(y2p, pwt, p_b, ofs);
    // deformable sampling + final GEMM + residual
    deform_gemm<<<NPIX / 16, 256, 0, stream>>>(y2p, dwt, ofs, x, (float*)d_out);
}
